// Transformer_17592186044419
// MI455X (gfx1250) — matrix operand order repaired and checked by reading, not compiled, not run on hardware
//
#include <hip/hip_runtime.h>

typedef unsigned short u16;
typedef __attribute__((ext_vector_type(8)))  __bf16 v8bf;
typedef __attribute__((ext_vector_type(16))) __bf16 v16bf;
typedef __attribute__((ext_vector_type(8)))  float  v8f;

#define HD     16
#define DH     64
#define LSEQ   1024
#define DMODEL 1024
#define BATCH  8

// ---------- helpers ----------

__device__ __forceinline__ u16 f32_to_bf16(float f) {
  unsigned u = __float_as_uint(f);
  u += 0x7FFFu + ((u >> 16) & 1u);   // round to nearest even
  return (u16)(u >> 16);
}

// A-fragment: 16(M) x 32(K) bf16.  lane 0-15: row=lane, K = kb..kb+7 and kb+16..kb+23
// (kb = 0 for lanes 0-15, 8 for lanes 16-31) -> two 16-byte loads per lane.
__device__ __forceinline__ v16bf load_frag_a(const u16* p0, int ld, int lane) {
  const u16* p = p0 + (size_t)(lane & 15) * ld + ((lane >> 4) << 3);
  v8bf lo = *(const v8bf*)(p);
  v8bf hi = *(const v8bf*)(p + 16);
  return __builtin_shufflevector(lo, hi, 0,1,2,3,4,5,6,7,8,9,10,11,12,13,14,15);
}

// B-fragment: 32(K) x 16(N) bf16, loaded from B^T stored [N,K] row-major.
// lane -> N = lane&15; K = kb..kb+7 and kb+16..kb+23 (kb = 8*(lane>>4)) -> two 16-byte loads.
__device__ __forceinline__ v16bf load_frag_b(const u16* p0, int ld, int lane) {
  const u16* p = p0 + (size_t)(lane & 15) * ld + ((lane >> 4) << 3);
  v8bf lo = *(const v8bf*)(p);
  v8bf hi = *(const v8bf*)(p + 16);
  return __builtin_shufflevector(lo, hi, 0,1,2,3,4,5,6,7,8,9,10,11,12,13,14,15);
}

__device__ __forceinline__ v8f wmma_bf16(v16bf a, v16bf b, v8f c) {
  return __builtin_amdgcn_wmma_f32_16x16x32_bf16(false, a, false, b, (short)0, c,
                                                 false, false);
}

// Async global -> LDS copy (CDNA5, ASYNCcnt-tracked). 16 bytes per lane.
// LDS offset = low 32 bits of the generic address of a __shared__ object
// (ISA: LDS aperture maps addr[31:0] to the DS offset).
__device__ __forceinline__ void async_copy_b128(const u16* g, u16* l) {
  unsigned lds = (unsigned)(size_t)l;
  unsigned long long ga = (unsigned long long)(size_t)g;
  asm volatile("global_load_async_to_lds_b128 %0, %1, off"
               :: "v"(lds), "v"(ga) : "memory");
}

template <int N>
__device__ __forceinline__ void wait_asynccnt() {
  asm volatile("s_wait_asynccnt %0" :: "i"(N) : "memory");
}

// ---------- conversion kernels ----------

__global__ void k_cvt_bf16(const float* __restrict__ in, u16* __restrict__ out, int n) {
  int i = (blockIdx.x * 256 + threadIdx.x) * 4;
  if (i + 3 < n) {
    float4 v = *(const float4*)(in + i);
    out[i + 0] = f32_to_bf16(v.x);
    out[i + 1] = f32_to_bf16(v.y);
    out[i + 2] = f32_to_bf16(v.z);
    out[i + 3] = f32_to_bf16(v.w);
  }
}

// W [K][N] fp32 row-major -> Wt [N][K] bf16 row-major, LDS-tiled for coalescing.
__global__ void k_transpose_cvt(const float* __restrict__ W, u16* __restrict__ Wt,
                                int K, int N) {
  __shared__ float tile[32][33];
  const int kt = blockIdx.x * 32, nt = blockIdx.y * 32;
  const int tx = threadIdx.x, ty = threadIdx.y;     // block (32,8)
  #pragma unroll
  for (int i = 0; i < 32; i += 8)
    tile[ty + i][tx] = W[(size_t)(kt + ty + i) * N + nt + tx];
  __syncthreads();
  #pragma unroll
  for (int i = 0; i < 32; i += 8)
    Wt[(size_t)(nt + ty + i) * K + kt + tx] = f32_to_bf16(tile[tx][ty + i]);
}

// V [B,L,D] bf16 -> Vt [B,H,DH,L] bf16 (coalesced on the write side)
__global__ void k_vtrans(const u16* __restrict__ v, u16* __restrict__ vt) {
  int idx = blockIdx.x * 256 + threadIdx.x;         // over B*H*DH*L = 8M
  int l  = idx & (LSEQ - 1);
  int d  = (idx >> 10) & (DH - 1);
  int bh = idx >> 16;
  int b = bh >> 4, h = bh & 15;
  vt[idx] = v[(size_t)(b * LSEQ + l) * DMODEL + h * DH + d];
}

// ---------- WMMA GEMM:  C[M,N] = act(A[M,K] @ Bt[N,K]^T + bias) ----------
// 256 threads = 8 waves; wave tile 32x32; block tile 128x64.
// Double-buffered async global->LDS staging: per K-stage of 32, the block stages
// A[128x32] (8KB) + Bt[64x32] (4KB) = 12KB; 3 async B128 ops per thread/stage.

template <int ACT, int OUTBF>
__global__ void __launch_bounds__(256)
k_gemm(const u16* __restrict__ A, const u16* __restrict__ Bt,
       const float* __restrict__ bias, u16* __restrict__ Cb,
       float* __restrict__ Cf, int M, int N, int K) {
  __shared__ u16 sA[2][128 * 32];
  __shared__ u16 sB[2][64 * 32];

  const int t    = threadIdx.x;
  const int lane = t & 31;
  const int wid  = t >> 5;
  const int bm_blk = blockIdx.x * 128;
  const int bn_blk = blockIdx.y * 64;
  const int wm = (wid >> 1) * 32;
  const int wn = (wid & 1) * 32;

  // this thread's copy lanes: A chunks t and t+256, B chunk t (16B each,
  // chunk c <-> row c>>2, col (c&3)*8 of a row-major [rows][32] u16 tile)
  const u16* gA0 = A  + (size_t)(bm_blk + (t >> 2)) * K + (t & 3) * 8;
  const u16* gA1 = A  + (size_t)(bm_blk + 64 + (t >> 2)) * K + (t & 3) * 8;
  const u16* gB  = Bt + (size_t)(bn_blk + (t >> 2)) * K + (t & 3) * 8;

  const int S = K / 32;
  auto issue = [&](int s, int buf) {
    const int kOff = s * 32;
    async_copy_b128(gA0 + kOff, &sA[buf][t * 8]);
    async_copy_b128(gA1 + kOff, &sA[buf][t * 8 + 2048]);
    async_copy_b128(gB  + kOff, &sB[buf][t * 8]);
  };

  issue(0, 0);
  v8f c00 = {}, c01 = {}, c10 = {}, c11 = {};
  for (int s = 0; s < S; ++s) {
    const int buf = s & 1;
    if (s + 1 < S) {
      issue(s + 1, buf ^ 1);
      wait_asynccnt<3>();     // ASYNCcnt in-order: stage s complete
    } else {
      wait_asynccnt<0>();
    }
    __syncthreads();
    v16bf a0 = load_frag_a(&sA[buf][wm * 32],        32, lane);
    v16bf a1 = load_frag_a(&sA[buf][(wm + 16) * 32], 32, lane);
    v16bf b0 = load_frag_b(&sB[buf][wn * 32],        32, lane);
    v16bf b1 = load_frag_b(&sB[buf][(wn + 16) * 32], 32, lane);
    c00 = wmma_bf16(a0, b0, c00);
    c01 = wmma_bf16(a0, b1, c01);
    c10 = wmma_bf16(a1, b0, c10);
    c11 = wmma_bf16(a1, b1, c11);
    __syncthreads();          // all waves done reading buf before it is reused
  }

  const int bm = bm_blk + wm, bn = bn_blk + wn;
  const int cc = lane & 15, rr = (lane >> 4) << 3;
  const float bias0 = bias[bn + cc];
  const float bias1 = bias[bn + 16 + cc];
  #pragma unroll
  for (int i = 0; i < 8; i++) {
    float v0 = c00[i] + bias0, v1 = c01[i] + bias1;
    float v2 = c10[i] + bias0, v3 = c11[i] + bias1;
    if (ACT == 1) {   // exact GELU
      v0 = 0.5f * v0 * (1.0f + erff(v0 * 0.70710678118654752f));
      v1 = 0.5f * v1 * (1.0f + erff(v1 * 0.70710678118654752f));
      v2 = 0.5f * v2 * (1.0f + erff(v2 * 0.70710678118654752f));
      v3 = 0.5f * v3 * (1.0f + erff(v3 * 0.70710678118654752f));
    }
    size_t r0 = (size_t)(bm + rr + i) * N;
    size_t r1 = (size_t)(bm + 16 + rr + i) * N;
    if (OUTBF) {
      Cb[r0 + bn + cc]      = f32_to_bf16(v0);
      Cb[r0 + bn + 16 + cc] = f32_to_bf16(v1);
      Cb[r1 + bn + cc]      = f32_to_bf16(v2);
      Cb[r1 + bn + 16 + cc] = f32_to_bf16(v3);
    } else {
      Cf[r0 + bn + cc]      = v0;
      Cf[r0 + bn + 16 + cc] = v1;
      Cf[r1 + bn + cc]      = v2;
      Cf[r1 + bn + 16 + cc] = v3;
    }
  }
}

// ---------- flash attention (online softmax) ----------
// grid (L/64, H, B), block 128 (4 waves). Each wave: 16 query rows, full Dh=64.

__global__ void __launch_bounds__(128)
k_attention(const u16* __restrict__ Q, const u16* __restrict__ Kk,
            const u16* __restrict__ Vt, float* __restrict__ O) {
  const int lane = threadIdx.x & 31;
  const int wid  = threadIdx.x >> 5;
  const int b = blockIdx.z, h = blockIdx.y;
  const int qrow = blockIdx.x * 64 + wid * 16;

  const u16* Qp = Q  + (size_t)b * LSEQ * DMODEL + h * DH;
  const u16* Kp = Kk + (size_t)b * LSEQ * DMODEL + h * DH;
  const u16* Vp = Vt + (size_t)(b * HD + h) * DH * LSEQ;

  v16bf q0 = load_frag_a(Qp + (size_t)qrow * DMODEL,      DMODEL, lane);
  v16bf q1 = load_frag_a(Qp + (size_t)qrow * DMODEL + 32, DMODEL, lane);

  float mi[8], li[8];
  #pragma unroll
  for (int i = 0; i < 8; i++) { mi[i] = -3.0e38f; li[i] = 0.0f; }
  v8f o0 = {}, o1 = {}, o2 = {}, o3 = {};

  __shared__ u16 psh[4][16 * 32];
  u16* pw = psh[wid];
  const int cc = lane & 15, rr = (lane >> 4) << 3;
  const float sc = 0.125f;                       // 1/sqrt(Dh)

  for (int j = 0; j < LSEQ; j += 32) {
    // S = Q Kj^T  (two 16x16 score tiles covering 32 keys)
    v16bf k0 = load_frag_b(Kp + (size_t)j * DMODEL,             DMODEL, lane);
    v16bf k1 = load_frag_b(Kp + (size_t)j * DMODEL + 32,        DMODEL, lane);
    v16bf k2 = load_frag_b(Kp + (size_t)(j + 16) * DMODEL,      DMODEL, lane);
    v16bf k3 = load_frag_b(Kp + (size_t)(j + 16) * DMODEL + 32, DMODEL, lane);
    v8f s0 = {}, s1 = {};
    s0 = wmma_bf16(q0, k0, s0);  s0 = wmma_bf16(q1, k1, s0);
    s1 = wmma_bf16(q0, k2, s1);  s1 = wmma_bf16(q1, k3, s1);

    float p0[8], p1[8];
    #pragma unroll
    for (int i = 0; i < 8; i++) {
      float a = s0[i] * sc, c = s1[i] * sc;
      float t = fmaxf(a, c);                    // row max over this key tile
      t = fmaxf(t, __shfl_xor(t, 1));
      t = fmaxf(t, __shfl_xor(t, 2));
      t = fmaxf(t, __shfl_xor(t, 4));
      t = fmaxf(t, __shfl_xor(t, 8));
      float mn    = fmaxf(mi[i], t);
      float alpha = __expf(mi[i] - mn);
      float e0 = __expf(a - mn);
      float e1 = __expf(c - mn);
      float rs = e0 + e1;                       // row sum
      rs += __shfl_xor(rs, 1);
      rs += __shfl_xor(rs, 2);
      rs += __shfl_xor(rs, 4);
      rs += __shfl_xor(rs, 8);
      li[i] = li[i] * alpha + rs;
      mi[i] = mn;
      o0[i] *= alpha; o1[i] *= alpha; o2[i] *= alpha; o3[i] *= alpha;
      p0[i] = e0; p1[i] = e1;
    }

    // C-layout probs -> row-major [16q][32k] in LDS -> reload as A-fragment
    #pragma unroll
    for (int i = 0; i < 8; i++) {
      pw[(rr + i) * 32 + cc]      = f32_to_bf16(p0[i]);
      pw[(rr + i) * 32 + cc + 16] = f32_to_bf16(p1[i]);
    }
    __syncthreads();
    v16bf pa = load_frag_a(pw, 32, lane);
    v16bf v0 = load_frag_b(Vp + (size_t)0  * LSEQ + j, LSEQ, lane);
    v16bf v1 = load_frag_b(Vp + (size_t)16 * LSEQ + j, LSEQ, lane);
    v16bf v2 = load_frag_b(Vp + (size_t)32 * LSEQ + j, LSEQ, lane);
    v16bf v3 = load_frag_b(Vp + (size_t)48 * LSEQ + j, LSEQ, lane);
    o0 = wmma_bf16(pa, v0, o0);
    o1 = wmma_bf16(pa, v1, o1);
    o2 = wmma_bf16(pa, v2, o2);
    o3 = wmma_bf16(pa, v3, o3);
    __syncthreads();
  }

  float* Op = O + (size_t)b * LSEQ * DMODEL + h * DH;
  #pragma unroll
  for (int i = 0; i < 8; i++) {
    float inv = 1.0f / li[i];
    size_t ro = (size_t)(qrow + rr + i) * DMODEL;
    Op[ro + 0  + cc] = o0[i] * inv;
    Op[ro + 16 + cc] = o1[i] * inv;
    Op[ro + 32 + cc] = o2[i] * inv;
    Op[ro + 48 + cc] = o3[i] * inv;
  }
}

// ---------- residual + LayerNorm (one block per row, D=1024, 256 thr) ----------

__global__ void __launch_bounds__(256)
k_add_ln(const float* __restrict__ X, const float* __restrict__ R,
         float* __restrict__ Of, u16* __restrict__ Ob, int write_bf) {
  const size_t row = blockIdx.x;
  const float* xp = X + row * DMODEL;
  const float* rp = R + row * DMODEL;
  const int lane = threadIdx.x & 31, wid = threadIdx.x >> 5;
  __shared__ float red[8];

  float v[4], s = 0.0f;
  #pragma unroll
  for (int i = 0; i < 4; i++) {
    int idx = threadIdx.x + i * 256;
    v[i] = xp[idx] + rp[idx];
    s += v[i];
  }
  for (int m = 16; m >= 1; m >>= 1) s += __shfl_xor(s, m);
  if (lane == 0) red[wid] = s;
  __syncthreads();
  s = red[0] + red[1] + red[2] + red[3] + red[4] + red[5] + red[6] + red[7];
  const float mu = s * (1.0f / DMODEL);

  float var = 0.0f;
  #pragma unroll
  for (int i = 0; i < 4; i++) { float d = v[i] - mu; var += d * d; }
  __syncthreads();
  for (int m = 16; m >= 1; m >>= 1) var += __shfl_xor(var, m);
  if (lane == 0) red[wid] = var;
  __syncthreads();
  var = (red[0] + red[1] + red[2] + red[3] + red[4] + red[5] + red[6] + red[7])
        * (1.0f / DMODEL);
  const float rstd = rsqrtf(var + 1e-5f);

  #pragma unroll
  for (int i = 0; i < 4; i++) {
    int idx = threadIdx.x + i * 256;
    float y = (v[i] - mu) * rstd;
    Of[row * DMODEL + idx] = y;
    if (write_bf) Ob[row * DMODEL + idx] = f32_to_bf16(y);
  }
}

// ---------- launcher ----------

extern "C" void kernel_launch(void* const* d_in, const int* in_sizes, int n_in,
                              void* d_out, int out_size, void* d_ws, size_t ws_size,
                              hipStream_t stream) {
  (void)in_sizes; (void)n_in; (void)out_size; (void)ws_size;
  const float* x  = (const float*)d_in[0];
  const float* Wq = (const float*)d_in[1];
  const float* bq = (const float*)d_in[2];
  const float* Wk = (const float*)d_in[3];
  const float* bk = (const float*)d_in[4];
  const float* Wv = (const float*)d_in[5];
  const float* bv = (const float*)d_in[6];
  const float* W1 = (const float*)d_in[7];
  const float* b1 = (const float*)d_in[8];
  const float* W2 = (const float*)d_in[9];
  const float* b2 = (const float*)d_in[10];
  float* out = (float*)d_out;

  char* w = (char*)d_ws;
  const size_t MB = 1024 * 1024;
  // Workspace layout (offsets in MB). q/k/v/vt region (38..102) is reused as
  // mid_bf (64MB) during the FFN; attn_f is reused as ffn_f; x_bf as h1_bf.
  u16*   x_bf   = (u16*)(w);                 // 16MB, reused as h1_bf
  u16*   wq_t   = (u16*)(w + 16 * MB);       //  2MB
  u16*   wk_t   = (u16*)(w + 18 * MB);       //  2MB
  u16*   wv_t   = (u16*)(w + 20 * MB);       //  2MB
  u16*   w1_t   = (u16*)(w + 22 * MB);       //  8MB
  u16*   w2_t   = (u16*)(w + 30 * MB);       //  8MB
  u16*   q_bf   = (u16*)(w + 38 * MB);       // 16MB
  u16*   k_bf   = (u16*)(w + 54 * MB);       // 16MB
  u16*   v_bf   = (u16*)(w + 70 * MB);       // 16MB
  u16*   vt_bf  = (u16*)(w + 86 * MB);       // 16MB
  float* attn_f = (float*)(w + 102 * MB);    // 32MB, reused as ffn_f
  float* h1_f   = (float*)(w + 134 * MB);    // 32MB  -> 166MB total
  u16*   mid_bf = q_bf;
  float* ffn_f  = attn_f;
  u16*   h1_bf  = x_bf;

  const int Mrows = BATCH * LSEQ;            // 8192
  const int D4 = 4 * DMODEL;                 // 4096

  // activations -> bf16
  k_cvt_bf16<<<dim3((Mrows * DMODEL) / 1024), 256, 0, stream>>>(x, x_bf, Mrows * DMODEL);

  // weights -> [N,K] bf16
  dim3 tb(32, 8);
  k_transpose_cvt<<<dim3(DMODEL / 32, DMODEL / 32), tb, 0, stream>>>(Wq, wq_t, DMODEL, DMODEL);
  k_transpose_cvt<<<dim3(DMODEL / 32, DMODEL / 32), tb, 0, stream>>>(Wk, wk_t, DMODEL, DMODEL);
  k_transpose_cvt<<<dim3(DMODEL / 32, DMODEL / 32), tb, 0, stream>>>(Wv, wv_t, DMODEL, DMODEL);
  k_transpose_cvt<<<dim3(DMODEL / 32, D4 / 32),     tb, 0, stream>>>(W1, w1_t, DMODEL, D4);
  k_transpose_cvt<<<dim3(D4 / 32, DMODEL / 32),     tb, 0, stream>>>(W2, w2_t, D4, DMODEL);

  // Q, K, V projections (bf16 out)
  dim3 gproj(Mrows / 128, DMODEL / 64);
  k_gemm<0, 1><<<gproj, 256, 0, stream>>>(x_bf, wq_t, bq, q_bf, nullptr, Mrows, DMODEL, DMODEL);
  k_gemm<0, 1><<<gproj, 256, 0, stream>>>(x_bf, wk_t, bk, k_bf, nullptr, Mrows, DMODEL, DMODEL);
  k_gemm<0, 1><<<gproj, 256, 0, stream>>>(x_bf, wv_t, bv, v_bf, nullptr, Mrows, DMODEL, DMODEL);

  // V -> [B,H,Dh,L]
  k_vtrans<<<dim3((Mrows * DMODEL) / 256), 256, 0, stream>>>(v_bf, vt_bf);

  // attention
  k_attention<<<dim3(LSEQ / 64, HD, BATCH), 128, 0, stream>>>(q_bf, k_bf, vt_bf, attn_f);

  // LN1: h1 = LN(x + attn)  (fp32 + bf16 copies)
  k_add_ln<<<dim3(Mrows), 256, 0, stream>>>(x, attn_f, h1_f, h1_bf, 1);

  // FFN
  k_gemm<1, 1><<<dim3(Mrows / 128, D4 / 64), 256, 0, stream>>>(h1_bf, w1_t, b1, mid_bf, nullptr, Mrows, D4, DMODEL);
  k_gemm<0, 0><<<gproj, 256, 0, stream>>>(mid_bf, w2_t, b2, nullptr, ffn_f, Mrows, DMODEL, D4);

  // LN2 -> output
  k_add_ln<<<dim3(Mrows), 256, 0, stream>>>(h1_f, ffn_f, out, nullptr, 0);
}
